// MultiStreamMambaEncoder_33638183862314
// MI455X (gfx1250) — matrix operand order repaired and checked by reading, not compiled, not run on hardware
//
#include <hip/hip_runtime.h>
#include <hip/hip_bf16.h>

typedef __bf16 bf16;
typedef __attribute__((ext_vector_type(16))) __bf16 v16bf;
typedef __attribute__((ext_vector_type(8)))  __bf16 v8bf;
typedef __attribute__((ext_vector_type(8)))  float  v8f;
typedef __attribute__((ext_vector_type(4)))  float  v4f;

// Model dims
#define B_SZ   64
#define T_SZ   64
#define N_AG   64
#define DM     128
#define DI     256
#define DS_N   16
#define DTR    8

// ---------------------------------------------------------------------------
// WMMA GEMM:  C[M,N] = A[M,K] @ Wt[N,K]^T (+bias) (+resid) (+optional relu)
// A fp32 row-major, Wt bf16 row-major [N,K] (i.e. B transposed).
// One wave per 16x16 output tile; grid.x == number of tiles, blockDim == 32,
// so EXEC is all-ones for every wave that reaches the WMMA (ISA requirement).
// Fragment layouts per CDNA5 ISA 7.12.2:
//   A 16x32 bf16 : lane<16 -> row M=lane,   K = kb+{0..7, 16..23}
//                  lane>=16-> row M=lane-16,K = kb+{8..15,24..31}
//   B 32x16 bf16 : lane<16 -> col N=lane,   K = kb+{0..7, 16..23}
//                  lane>=16-> col N=lane-16,K = kb+{8..15,24..31}
//   C 16x16 f32  : VGPR i -> row tm + half*8 + i, col tn + (lane&15)
// ---------------------------------------------------------------------------
__global__ void k_wmma_gemm(const float* __restrict__ A, const bf16* __restrict__ Wt,
                            const float* __restrict__ bias, const float* __restrict__ resid,
                            float* __restrict__ C, int M, int N, int K, int act)
{
  const int tilesN = N >> 4;
  const int wave   = blockIdx.x;
  const int tm     = (wave / tilesN) << 4;
  const int tn     = (wave % tilesN) << 4;
  if (tm >= M) return;                       // wave-uniform guard (never taken: grid==tiles)
  const int lane = threadIdx.x & 31;
  const int half = lane >> 4;
  const int l    = lane & 15;

  const float* arow = A  + (size_t)(tm + l) * K;
  const bf16*  brow = Wt + (size_t)(tn + l) * K;

  v8f acc = {};
  for (int kb = 0; kb < K; kb += 32) {
    // A fragment: two contiguous 8-float chunks, convert fp32 -> bf16 in regs
    const v4f* pa0 = (const v4f*)(arow + kb + half * 8);
    const v4f* pa1 = (const v4f*)(arow + kb + 16 + half * 8);
    v4f a0 = pa0[0], a1 = pa0[1], a2 = pa1[0], a3 = pa1[1];
    v16bf af;
#pragma unroll
    for (int i = 0; i < 4; ++i) {
      af[i]      = (bf16)a0[i];
      af[4 + i]  = (bf16)a1[i];
      af[8 + i]  = (bf16)a2[i];
      af[12 + i] = (bf16)a3[i];
    }
    // B fragment: two 8-bf16 chunks from the transposed weight row (same K as A)
    const v8bf* pb0 = (const v8bf*)(brow + kb + half * 8);
    const v8bf* pb1 = (const v8bf*)(brow + kb + 16 + half * 8);
    v8bf b0 = pb0[0], b1 = pb1[0];
    v16bf bfv;
#pragma unroll
    for (int i = 0; i < 8; ++i) { bfv[i] = b0[i]; bfv[8 + i] = b1[i]; }

    acc = __builtin_amdgcn_wmma_f32_16x16x32_bf16(false, af, false, bfv,
                                                  (short)0, acc, false, false);
  }

#pragma unroll
  for (int i = 0; i < 8; ++i) {
    int r = tm + (half << 3) + i;
    int c = tn + l;
    float v = acc[i];
    if (bias)  v += bias[c];
    if (resid) v += resid[(size_t)r * N + c];
    if (act == 1) v = fmaxf(v, 0.f);
    C[(size_t)r * N + c] = v;
  }
}

// Transpose-convert weights: src fp32 [K,N] -> dst bf16 [Npad,K] (zero pad rows N..Npad)
__global__ void k_tcvt(const float* __restrict__ src, bf16* __restrict__ dst,
                       int K, int N, int Npad)
{
  int i = blockIdx.x * blockDim.x + threadIdx.x;
  if (i >= Npad * K) return;
  int n = i / K, k = i - n * K;
  float v = (n < N) ? src[(size_t)k * N + n] : 0.f;
  dst[(size_t)n * K + k] = (bf16)v;
}

// RMSNorm over last dim (128). One token per 128-thread block, LDS tree reduce.
__global__ void k_rmsnorm(const float* __restrict__ x, const float* __restrict__ w,
                          float* __restrict__ out, int M)
{
  int m = blockIdx.x, d = threadIdx.x;
  float v = x[(size_t)m * DM + d];
  __shared__ float red[DM];
  red[d] = v * v;
  __syncthreads();
  for (int s = 64; s > 0; s >>= 1) {
    if (d < s) red[d] += red[d + s];
    __syncthreads();
  }
  float r = rsqrtf(red[0] * (1.f / DM) + 1e-5f);
  out[(size_t)m * DM + d] = v * r * w[d];
  (void)M;
}

// Poly front: h = relu( LN( x @ w1 + b1 ) ). One token per block (128 threads).
__global__ void k_poly_front(const float* __restrict__ X, int in_dim,
                             const float* __restrict__ w1, const float* __restrict__ b1,
                             const float* __restrict__ lnw, const float* __restrict__ lnb,
                             float* __restrict__ out)
{
  int m = blockIdx.x, d = threadIdx.x;
  const float* x = X + (size_t)m * in_dim;
  float v = b1[d];
  for (int k = 0; k < in_dim; ++k) v += x[k] * w1[k * DM + d];
  __shared__ float s1[DM], s2[DM];
  s1[d] = v; s2[d] = v * v;
  __syncthreads();
  for (int s = 64; s > 0; s >>= 1) {
    if (d < s) { s1[d] += s1[d + s]; s2[d] += s2[d + s]; }
    __syncthreads();
  }
  float mu  = s1[0] * (1.f / DM);
  float var = s2[0] * (1.f / DM) - mu * mu;
  float h = (v - mu) * rsqrtf(var + 1e-5f) * lnw[d] + lnb[d];
  out[(size_t)m * DM + d] = fmaxf(h, 0.f);
}

// Depthwise causal conv (width 4) over time + SiLU. Reads xc = xz[:, :256].
__global__ void k_conv_silu(const float* __restrict__ xz, const float* __restrict__ cw,
                            const float* __restrict__ cb, float* __restrict__ out,
                            int M, int L)
{
  int i = blockIdx.x * blockDim.x + threadIdx.x;
  if (i >= M * DI) return;
  int m = i >> 8, c = i & (DI - 1);
  int t = m % L;
  float acc = cb[c];
#pragma unroll
  for (int j = 0; j < 4; ++j) {
    int tt = t - 3 + j;
    if (tt >= 0) acc += xz[(size_t)(m - 3 + j) * (2 * DI) + c] * cw[c * 4 + j];
  }
  out[(size_t)m * DI + c] = acc / (1.f + __expf(-acc));   // silu
}

// delta = softplus( dt @ dt_w + dt_b ),  dt = dbc[:, 0:8]  (dbc padded to 48 cols)
__global__ void k_dtproj(const float* __restrict__ dbc, const float* __restrict__ dtw,
                         const float* __restrict__ dtb, float* __restrict__ delta, int M)
{
  int i = blockIdx.x * blockDim.x + threadIdx.x;
  if (i >= M * DI) return;
  int m = i >> 8, c = i & (DI - 1);
  float acc = dtb[c];
  const float* row = dbc + (size_t)m * 48;
#pragma unroll
  for (int r = 0; r < DTR; ++r) acc += row[r] * dtw[r * DI + c];
  float sp = (acc > 20.f) ? acc : log1pf(__expf(acc));
  delta[(size_t)m * DI + c] = sp;
}

// Selective scan: one thread per (batch, channel); 16-state recurrence over time.
// y = (scan) + xc*D, then y *= silu(gate). Bm = dbc[:,8:24], Cm = dbc[:,24:40].
__global__ void k_scan(const float* __restrict__ delta, const float* __restrict__ dbc,
                       const float* __restrict__ xcs, const float* __restrict__ xz,
                       const float* __restrict__ A_log, const float* __restrict__ Dp,
                       float* __restrict__ ys, int Bn, int L)
{
  int i = blockIdx.x * blockDim.x + threadIdx.x;
  if (i >= Bn * DI) return;
  int b = i >> 8, c = i & (DI - 1);
  float Acs[DS_N], hs[DS_N];
#pragma unroll
  for (int s = 0; s < DS_N; ++s) { Acs[s] = -__expf(A_log[c * DS_N + s]); hs[s] = 0.f; }
  float Dc = Dp[c];
  for (int t = 0; t < L; ++t) {
    size_t m = (size_t)b * L + t;
    float d  = delta[m * DI + c];
    float xv = xcs[m * DI + c];
    const float* db = dbc + m * 48;
    float y = 0.f;
#pragma unroll
    for (int s = 0; s < DS_N; ++s) {
      hs[s] = __expf(d * Acs[s]) * hs[s] + d * db[8 + s] * xv;
      y += hs[s] * db[24 + s];
    }
    y += xv * Dc;
    float g = xz[m * (2 * DI) + DI + c];
    y *= g / (1.f + __expf(-g));                 // * silu(gate)
    ys[m * DI + c] = y;
  }
}

// Mean over sequence dim: feat[b,d] = mean_t src[(b*L+t), d]
__global__ void k_mean(const float* __restrict__ src, float* __restrict__ feat, int Bn, int L)
{
  int i = blockIdx.x * blockDim.x + threadIdx.x;
  if (i >= Bn * DM) return;
  int b = i >> 7, d = i & (DM - 1);
  float s = 0.f;
  for (int t = 0; t < L; ++t) s += src[((size_t)b * L + t) * DM + d];
  feat[i] = s * (1.f / (float)L);
}

// Max-reduce over middle dim: src[G,R,128] -> dst rows remapped
// (outer = g/rowsPer, inner = g%rowsPer, dstRow = outer*rowStride + inner)
__global__ void k_maxred(const float* __restrict__ src, float* __restrict__ dst,
                         int G, int R, int rowsPer, int rowStride)
{
  int i = blockIdx.x * blockDim.x + threadIdx.x;
  if (i >= G * DM) return;
  int g = i >> 7, d = i & (DM - 1);
  float mx = -3.402823e38f;
  for (int r = 0; r < R; ++r)
    mx = fmaxf(mx, src[((size_t)g * R + r) * DM + d]);
  int outer = g / rowsPer, inner = g - outer * rowsPer;
  dst[((size_t)outer * rowStride + inner) * DM + d] = mx;
}

// Small-K embedding: out[m,d] = b[d] + sum_k x[m,k]*w[k,d]
__global__ void k_embed(const float* __restrict__ x, const float* __restrict__ w,
                        const float* __restrict__ bias, float* __restrict__ out,
                        int M, int in_dim)
{
  int i = blockIdx.x * blockDim.x + threadIdx.x;
  if (i >= M * DM) return;
  int m = i >> 7, d = i & (DM - 1);
  float acc = bias[d];
  for (int k = 0; k < in_dim; ++k) acc += x[(size_t)m * in_dim + k] * w[k * DM + d];
  out[(size_t)m * DM + d] = acc;
}

// traffic tokens into rows 50..55 of each 56-row scene sequence
__global__ void k_traf(const float* __restrict__ traf, const float* __restrict__ w,
                       const float* __restrict__ bias, float* __restrict__ seq, int Bn)
{
  int i = blockIdx.x * blockDim.x + threadIdx.x;
  if (i >= Bn * 6 * DM) return;
  int d = i & (DM - 1);
  int j = (i >> 7) % 6;
  int b = i / (6 * DM);
  seq[((size_t)(b * 56 + 50 + j)) * DM + d] = traf[b * 6 + j] * w[d] + bias[d];
}

// stack three [B,128] features into [B,3,128]
__global__ void k_stack3(const float* __restrict__ a, const float* __restrict__ b,
                         const float* __restrict__ c, float* __restrict__ out, int Bn)
{
  int i = blockIdx.x * blockDim.x + threadIdx.x;
  if (i >= Bn * DM) return;
  int bb = i >> 7, d = i & (DM - 1);
  out[((size_t)bb * 3 + 0) * DM + d] = a[i];
  out[((size_t)bb * 3 + 1) * DM + d] = b[i];
  out[((size_t)bb * 3 + 2) * DM + d] = c[i];
}

// ---------------------------------------------------------------------------
// Host side
// ---------------------------------------------------------------------------
struct MP {
  const float *A_log, *D, *conv_b, *conv_w, *dt_b, *dt_w, *in_w, *norm_w, *out_w, *x_proj_w, *norm_f;
};

static const size_t OF_WT    = 0;                                   // 2 MiB bf16 weights
static const size_t OF_BUFA  = (size_t)2 << 20;
static const size_t OF_BUFB  = OF_BUFA + (size_t)262144 * 128 * 4;  // 128 MiB each
static const size_t OF_SEQX  = OF_BUFB + (size_t)262144 * 128 * 4;
static const size_t OF_HRMS  = OF_SEQX + (size_t)4096 * 128 * 4;
static const size_t OF_XZ    = OF_HRMS + (size_t)4096 * 128 * 4;
static const size_t OF_XCS   = OF_XZ   + (size_t)4096 * 512 * 4;
static const size_t OF_DBC   = OF_XCS  + (size_t)4096 * 256 * 4;
static const size_t OF_DELTA = OF_DBC  + (size_t)4096 * 48 * 4;
static const size_t OF_YS    = OF_DELTA+ (size_t)4096 * 256 * 4;
static const size_t OF_EF    = OF_YS   + (size_t)4096 * 256 * 4;
static const size_t OF_SF    = OF_EF + 65536;
static const size_t OF_MF    = OF_SF + 65536;

static const size_t ENC_STRIDE = 221184;  // bf16 elems per encoder (2 layers)
static const size_t LAY_STRIDE = 110592;  // bf16 elems per layer: 65536 + 12288 + 32768

extern "C" void kernel_launch(void* const* d_in, const int* in_sizes, int n_in,
                              void* d_out, int out_size, void* d_ws, size_t ws_size,
                              hipStream_t stream) {
  (void)in_sizes; (void)n_in; (void)out_size; (void)ws_size;
  auto f = [&](int i) { return (const float*)d_in[i]; };

  char*  ws    = (char*)d_ws;
  bf16*  WT    = (bf16*)(ws + OF_WT);
  float* bufA  = (float*)(ws + OF_BUFA);
  float* bufB  = (float*)(ws + OF_BUFB);
  float* seqX  = (float*)(ws + OF_SEQX);
  float* hrms  = (float*)(ws + OF_HRMS);
  float* xz    = (float*)(ws + OF_XZ);
  float* xcs   = (float*)(ws + OF_XCS);
  float* dbc   = (float*)(ws + OF_DBC);
  float* delta = (float*)(ws + OF_DELTA);
  float* ys    = (float*)(ws + OF_YS);
  float* efeat = (float*)(ws + OF_EF);
  float* sfeat = (float*)(ws + OF_SF);
  float* mfeat = (float*)(ws + OF_MF);

  // params pytree leaves in sorted-key order (jax dict flatten):
  // 0 ego_hist  1 social_agents  2 map_scene  3 traf
  // 4 ego_b  5 ego_w
  // 6..16 gm (A_log,D,conv_b,conv_w,dt_b,dt_w,in_w,norm_w,out_w,x_proj_w,norm_f)
  // 17..22 map poly (b1,b2,ln_b,ln_w,w1,w2)
  // 23..33 sm   34..39 social poly   40..50 tm
  // 51 traf_b  52 traf_w   53..63 trm
  auto mk = [&](int i0) -> MP {
    return { f(i0), f(i0+1), f(i0+2), f(i0+3), f(i0+4), f(i0+5),
             f(i0+6), f(i0+7), f(i0+8), f(i0+9), f(i0+10) };
  };
  MP gm = mk(6), sm = mk(23), tm = mk(40), trm = mk(53);
  const MP encs[4] = { tm, trm, sm, gm };   // weight-region order

  auto tcvt = [&](const float* src, bf16* dst, int K, int N, int Npad) {
    int n = Npad * K;
    k_tcvt<<<(n + 255) / 256, 256, 0, stream>>>(src, dst, K, N, Npad);
  };
  auto gemm = [&](const float* A, const bf16* Wt, const float* bias, const float* resid,
                  float* C, int M, int N, int K) {
    int tiles = (M >> 4) * (N >> 4);
    k_wmma_gemm<<<tiles, 32, 0, stream>>>(A, Wt, bias, resid, C, M, N, K, 0);
  };

  // ---- prep: fp32 [K,N] weights -> bf16 [Npad,K] transposed ----
  for (int e = 0; e < 4; ++e)
    for (int l = 0; l < 2; ++l) {
      bf16* base = WT + (size_t)e * ENC_STRIDE + (size_t)l * LAY_STRIDE;
      tcvt(encs[e].in_w     + (size_t)l * 128 * 512, base,          128, 512, 512);
      tcvt(encs[e].x_proj_w + (size_t)l * 256 * 40,  base + 65536,  256, 40,  48);
      tcvt(encs[e].out_w    + (size_t)l * 256 * 128, base + 77824,  256, 128, 128);
    }
  tcvt(f(39), WT + 884736, 128, 128, 128);  // social.w2
  tcvt(f(22), WT + 901120, 128, 128, 128);  // map.w2

  // ---- mamba encoder driver ----
  auto run_mamba = [&](const MP& P, int encIdx, float* X, int Bn, int L, float* feat) {
    int M = Bn * L;
    for (int l = 0; l < 2; ++l) {
      const bf16* base = WT + (size_t)encIdx * ENC_STRIDE + (size_t)l * LAY_STRIDE;
      k_rmsnorm<<<M, 128, 0, stream>>>(X, P.norm_w + l * 128, hrms, M);
      gemm(hrms, base, nullptr, nullptr, xz, M, 512, 128);                     // in_proj
      { int n = M * DI;
        k_conv_silu<<<(n + 255) / 256, 256, 0, stream>>>(xz, P.conv_w + l * 1024,
                                                         P.conv_b + l * 256, xcs, M, L); }
      gemm(xcs, base + 65536, nullptr, nullptr, dbc, M, 48, 256);              // x_proj
      { int n = M * DI;
        k_dtproj<<<(n + 255) / 256, 256, 0, stream>>>(dbc, P.dt_w + l * 2048,
                                                      P.dt_b + l * 256, delta, M); }
      { int n = Bn * DI;
        k_scan<<<(n + 255) / 256, 256, 0, stream>>>(delta, dbc, xcs, xz,
                                                    P.A_log + l * 4096, P.D + l * 256,
                                                    ys, Bn, L); }
      gemm(ys, base + 77824, nullptr, X, X, M, 128, 256);                      // out_proj + residual
    }
    k_rmsnorm<<<M, 128, 0, stream>>>(X, P.norm_f, hrms, M);
    { int n = Bn * DM;
      k_mean<<<(n + 127) / 128, 128, 0, stream>>>(hrms, feat, Bn, L); }
  };

  // ---- 1) ego stream: embed -> tm -> e_feat ----
  { int n = 4096 * DM;
    k_embed<<<(n + 255) / 256, 256, 0, stream>>>(f(0), f(5), f(4), seqX, 4096, 6); }
  run_mamba(tm, 0, seqX, B_SZ, T_SZ, efeat);

  // ---- 2) social stream: poly (262144 tokens) -> max over T -> trm -> s_feat ----
  k_poly_front<<<262144, 128, 0, stream>>>(f(1), 6, f(38), f(34), f(37), f(36), bufA);
  gemm(bufA, WT + 884736, f(35), nullptr, bufB, 262144, 128, 128);
  { int n = 4096 * DM;
    k_maxred<<<(n + 255) / 256, 256, 0, stream>>>(bufB, seqX, 4096, 64, 64, 64); }
  run_mamba(trm, 1, seqX, B_SZ, N_AG, sfeat);

  // ---- 3) scene stream: map poly (32000 tokens) -> max over PTS -> +traf -> sm ----
  k_poly_front<<<32000, 128, 0, stream>>>(f(2), 3, f(21), f(17), f(20), f(19), bufA);
  gemm(bufA, WT + 901120, f(18), nullptr, bufB, 32000, 128, 128);
  { int n = 3200 * DM;
    k_maxred<<<(n + 255) / 256, 256, 0, stream>>>(bufB, seqX, 3200, 10, 50, 56); }
  { int n = B_SZ * 6 * DM;
    k_traf<<<(n + 255) / 256, 256, 0, stream>>>(f(3), f(52), f(51), seqX, B_SZ); }
  run_mamba(sm, 2, seqX, B_SZ, 56, mfeat);

  // ---- 4) fusion: stack 3 feats -> gm -> mean -> output [64,128] fp32 ----
  { int n = B_SZ * DM;
    k_stack3<<<(n + 255) / 256, 256, 0, stream>>>(efeat, sfeat, mfeat, seqX, B_SZ); }
  run_mamba(gm, 3, seqX, B_SZ, 3, (float*)d_out);
}
